// LinearAttention_20598663152021
// MI455X (gfx1250) — matrix operand order repaired and checked by reading, not compiled, not run on hardware
//
#include <hip/hip_runtime.h>

// ---------------------------------------------------------------------------
// Types / helpers for CDNA5 WMMA (wave32, v_wmma_f32_16x16x32_bf16)
// ---------------------------------------------------------------------------
typedef __attribute__((ext_vector_type(16))) __bf16 v16bf;
typedef __attribute__((ext_vector_type(8)))  float  v8f;

union BF16x16 { v16bf v; unsigned short u[16]; unsigned int d[8]; };
union F32x8   { v8f v; float f[8]; };

__device__ __forceinline__ unsigned short f2bf(float f) {
  unsigned int x = __builtin_bit_cast(unsigned int, f);
  x += 0x7fffu + ((x >> 16) & 1u);           // round-to-nearest-even
  return (unsigned short)(x >> 16);
}

// Pack two fp32 -> bf16x2 dword (v_cvt_pk_bf16_f32 when available).
__device__ __forceinline__ unsigned int pack_bf16(float lo, float hi) {
#if __has_builtin(__builtin_amdgcn_cvt_pk_bf16_f32)
  auto r = __builtin_amdgcn_cvt_pk_bf16_f32(lo, hi);
  return __builtin_bit_cast(unsigned int, r);
#else
  return (unsigned int)f2bf(lo) | ((unsigned int)f2bf(hi) << 16);
#endif
}

// Async global->LDS copy (gfx1250 GLOBAL_LOAD_ASYNC_TO_LDS_B128, ASYNCcnt).
// Probe-learned signature: (v4i addrspace(1)*, v4i addrspace(3)*, imm, cpol).
#if __has_builtin(__builtin_amdgcn_global_load_async_to_lds_b128)
#define HAVE_ASYNC_LDS 1
typedef __attribute__((__vector_size__(4 * sizeof(int)))) int async_v4i;
__device__ __forceinline__ void async_copy_b128(const void* g, void* l) {
  __builtin_amdgcn_global_load_async_to_lds_b128(
      (__attribute__((address_space(1))) async_v4i*)(unsigned long long)g,
      (__attribute__((address_space(3))) async_v4i*)(unsigned int)(unsigned long long)l,
      0, 0);
}
#define ASYNC_WAIT() asm volatile("s_wait_asynccnt 0x0" ::: "memory")
#else
#define HAVE_ASYNC_LDS 0
#define ASYNC_WAIT()
#endif

__device__ __forceinline__ v8f wmma_bf16(v16bf a, v16bf b, v8f c) {
  return __builtin_amdgcn_wmma_f32_16x16x32_bf16(
      false, a, false, b, (short)0, c, false, false);
}

// A-matrix 16x32 bf16: lane m=lane&15 holds row m; half=lane>>4;
// VGPR v holds K = 2v + (v>=4 ? 8 : 0) + 8*half and K+1 (contiguous dword).
__device__ __forceinline__ v16bf load_A(const unsigned short* base, int pitch,
                                        int row0, int k0, int lane) {
  BF16x16 t; int half = lane >> 4, m = lane & 15;
  const unsigned short* r = base + (row0 + m) * pitch + k0 + half * 8;
#pragma unroll
  for (int v = 0; v < 8; v++) {
    int k = 2 * v + (v >= 4 ? 8 : 0);
    t.d[v] = *(const unsigned int*)(r + k);
  }
  return t.v;
}

// B-matrix 32x16 bf16 where B[k][n] = Mt[n][k] (transposed source in LDS):
// lane n=lane&15 holds column n; VGPR v holds K = 2v + (v>=4 ? 8 : 0) + 8*half, K+1 (dword).
__device__ __forceinline__ v16bf load_B_trans(const unsigned short* base, int pitch,
                                              int n0, int k0, int lane) {
  BF16x16 t; int half = lane >> 4, n = lane & 15;
  const unsigned short* r = base + (n0 + n) * pitch + k0 + half * 8;
#pragma unroll
  for (int v = 0; v < 8; v++) t.d[v] = *(const unsigned int*)(r + 2 * v + (v >= 4 ? 8 : 0));
  return t.v;
}

// ---------------------------------------------------------------------------
// Problem constants
// ---------------------------------------------------------------------------
#define BB 4
#define NN 2048
#define DD 1024
#define HH 8
#define DK 128
#define DV 128
#define DF 256
#define CK 64
#define BN (BB * NN)   // 8192
#define DVS 64

// ---------------------------------------------------------------------------
// Prep kernels
// ---------------------------------------------------------------------------
__global__ void cvt_bf16_kernel(const float* __restrict__ in,
                                unsigned int* __restrict__ out, int npairs) {
  for (int i = blockIdx.x * blockDim.x + threadIdx.x; i < npairs;
       i += gridDim.x * blockDim.x)
    out[i] = pack_bf16(in[2 * i], in[2 * i + 1]);
}

// Wtilde[h*128+e][c] = sum_d fm[e][d] * W[h*128+d][c]  (hedgehog folded into proj)
__global__ void make_wtilde_kernel(const float* __restrict__ fm,
                                   const float* __restrict__ W,
                                   unsigned short* __restrict__ out) {
  int idx = blockIdx.x * 256 + threadIdx.x;      // 0 .. 512K-1 (pairs of cols)
  int r = idx >> 9, cp = (idx & 511) * 2;
  int h = r >> 7, e = r & 127;
  float s0 = 0.f, s1 = 0.f;
#pragma unroll 8
  for (int d = 0; d < DK; d++) {
    float f = fm[e * DK + d];
    const float* w = &W[(size_t)(h * DK + d) * DD + cp];
    s0 += f * w[0];
    s1 += f * w[1];
  }
  *(unsigned int*)&out[(size_t)r * DD + cp] = pack_bf16(s0, s1);
}

// ---------------------------------------------------------------------------
// Double-buffered tiled bf16 GEMM:
// C[M=8192,N=1024] = A[8192,1024] @ W^T, W row-major [1024,1024].
// MODE 0: hedgehog-q epilogue -> PhiQ bf16 [B,H,N,DF], feature pairs
//         interleaved as [exp(y), exp(-y)] (DF permutation, contraction-safe)
// MODE 1: hedgehog-k epilogue (-> PhiK)
// MODE 2: plain bf16 store per-head layout (-> Vh [B,H,N,DV])
// MODE 3: plain fp32 store (-> d_out [B,N,D])
// ---------------------------------------------------------------------------
template <int MODE>
__global__ __launch_bounds__(256) void gemm_bf16_kernel(
    const unsigned short* __restrict__ A, const unsigned short* __restrict__ Wb,
    const float* __restrict__ bias, unsigned short* __restrict__ outU,
    float* __restrict__ outF, float scale) {
  __shared__ unsigned short As[2][128 * 40];   // 128 rows x 32 k (+8 pad)
  __shared__ unsigned short Bs[2][128 * 40];

  const int tid = threadIdx.x, lane = tid & 31, wid = tid >> 5;
  const int wave_m = wid >> 1, wave_n = wid & 1;
  const int m0 = blockIdx.x * 128, n0 = blockIdx.y * 128;
  const int rowL = tid >> 2, seg = (tid & 3) * 8;   // this thread's stage slot

  const unsigned short* gA0 = A + (size_t)(m0 + rowL) * DD + seg;
  const unsigned short* gA1 = A + (size_t)(m0 + rowL + 64) * DD + seg;
  const unsigned short* gB0 = Wb + (size_t)(n0 + rowL) * DD + seg;
  const unsigned short* gB1 = Wb + (size_t)(n0 + rowL + 64) * DD + seg;

  // prologue: stage k-step 0 into buffer 0
  int4 ra0 = *(const int4*)gA0, ra1 = *(const int4*)gA1;
  int4 rb0 = *(const int4*)gB0, rb1 = *(const int4*)gB1;
  *(int4*)&As[0][rowL * 40 + seg] = ra0;
  *(int4*)&As[0][(rowL + 64) * 40 + seg] = ra1;
  *(int4*)&Bs[0][rowL * 40 + seg] = rb0;
  *(int4*)&Bs[0][(rowL + 64) * 40 + seg] = rb1;
  __syncthreads();

  F32x8 acc[2][4];
#pragma unroll
  for (int mt = 0; mt < 2; mt++)
#pragma unroll
    for (int nt = 0; nt < 4; nt++)
#pragma unroll
      for (int r = 0; r < 8; r++) acc[mt][nt].f[r] = 0.f;

  for (int kk = 0; kk < 32; kk++) {
    const int cur = kk & 1;
    const bool has_next = kk < 31;
    if (has_next) {                       // issue next tile's global loads early
      int k0 = (kk + 1) * 32;
      ra0 = *(const int4*)(gA0 + k0);
      ra1 = *(const int4*)(gA1 + k0);
      rb0 = *(const int4*)(gB0 + k0);
      rb1 = *(const int4*)(gB1 + k0);
    }
    v16bf a[2];
#pragma unroll
    for (int mt = 0; mt < 2; mt++)
      a[mt] = load_A(As[cur], 40, wave_m * 32 + mt * 16, 0, lane);
#pragma unroll
    for (int nt = 0; nt < 4; nt++) {
      v16bf b = load_B_trans(Bs[cur], 40, wave_n * 64 + nt * 16, 0, lane);
#pragma unroll
      for (int mt = 0; mt < 2; mt++)
        acc[mt][nt].v = wmma_bf16(a[mt], b, acc[mt][nt].v);
    }
    if (has_next) {                       // ping-pong store to the other buffer
      int nxt = cur ^ 1;
      *(int4*)&As[nxt][rowL * 40 + seg] = ra0;
      *(int4*)&As[nxt][(rowL + 64) * 40 + seg] = ra1;
      *(int4*)&Bs[nxt][rowL * 40 + seg] = rb0;
      *(int4*)&Bs[nxt][(rowL + 64) * 40 + seg] = rb1;
    }
    __syncthreads();
  }

  // Epilogue (C-layout: lane&15 = column, VGPR r -> row r + 8*(lane>>4))
  const int half = lane >> 4, nn = lane & 15;
#pragma unroll
  for (int mt = 0; mt < 2; mt++) {
#pragma unroll
    for (int nt = 0; nt < 4; nt++) {
      int rowt = m0 + wave_m * 32 + mt * 16;
      int c = n0 + wave_n * 64 + nt * 16 + nn;
#pragma unroll
      for (int r = 0; r < 8; r++) {
        int m = rowt + r + 8 * half;
        float y = acc[mt][nt].f[r];
        if constexpr (MODE <= 1) {
          y += bias[c & 127];
          int h = c >> 7, e = c & 127;
          int b_ = m >> 11, ntok = m & (NN - 1);
          size_t base = ((size_t)(b_ * HH + h) * NN + ntok) * DF;
          float ey = __expf(y), em = __expf(-y);
          *(unsigned int*)&outU[base + 2 * e] = pack_bf16(ey * scale, em * scale);
        } else if constexpr (MODE == 2) {
          int h = c >> 7, e = c & 127;
          int b_ = m >> 11, ntok = m & (NN - 1);
          outU[((size_t)(b_ * HH + h) * NN + ntok) * DV + e] = f2bf(y);
        } else {
          outF[(size_t)m * DD + c] = y;
        }
      }
    }
  }
}

// ---------------------------------------------------------------------------
// Chunked causal linear attention. grid = (B*H, 2 DV-slices of 64).
// All products oriented so WMMA D tiles store contiguously and all operands
// are contiguous-dword LDS reads. Per wave, both owned tiles share the same
// A operand (row tile = wid>>1), so A is loaded once per k-step.
//   Phase A: attn^T = Kc Qc^T  -> packed b128 stores into Asc[c][e]
//   Phase B: O^T = V^T attn^T + S^T Q^T -> contiguous global b128 stores
//   Phase C: S += Kc^T Vc (state in VGPRs) -> packed b128 mirror Sbt[v][d]
// Q chunk staging goes memory->LDS via GLOBAL_LOAD_ASYNC_TO_LDS_B128 when
// the toolchain exposes it (no VGPR transit; tracked by ASYNCcnt).
// ---------------------------------------------------------------------------
__global__ __launch_bounds__(256) void attn_kernel(
    const unsigned short* __restrict__ phiQ, const unsigned short* __restrict__ phiK,
    const unsigned short* __restrict__ Vh, float* __restrict__ Oacc) {
  __shared__ unsigned short Qs[64 * 264];    // 64 x 256 bf16 (+8 pad)
  __shared__ unsigned short Ks[64 * 264];
  __shared__ unsigned short Kt[256 * 72];    // Kt[d][c] transposed K
  __shared__ unsigned short Vt[64 * 72];     // Vt[v][c] transposed V slice
  __shared__ unsigned short Asc[64 * 72];    // Asc[c][e] masked scores, bf16
  __shared__ unsigned short Sbt[64 * 264];   // Sbt[v][d] = bf16(S[d][v])

  const int bh = blockIdx.x;                 // 0..31
  const int vs = blockIdx.y;                 // 0..1
  const int tid = threadIdx.x, lane = tid & 31, wid = tid >> 5;
  const int half = lane >> 4, nn = lane & 15;
  const int rt = wid >> 1;                   // wave's row-tile (et / vt)
  const int ct0 = (wid & 1) * 2;             // wave's first column tile

  for (int i = tid; i < 64 * 264; i += 256) Sbt[i] = 0;

  F32x8 Sreg[8];                             // this wave's state slice S[d][v]
#pragma unroll
  for (int i = 0; i < 8; i++)
#pragma unroll
    for (int r = 0; r < 8; r++) Sreg[i].f[r] = 0.f;

  const unsigned short* qb = phiQ + (size_t)bh * NN * DF;
  const unsigned short* kb = phiK + (size_t)bh * NN * DF;
  const unsigned short* vb = Vh + (size_t)bh * NN * DV + vs * DVS;
  float* ob = Oacc + (size_t)bh * NN * DV + vs * DVS;
  __syncthreads();

  for (int c0 = 0; c0 < NN; c0 += CK) {
    // Stage Qc (async direct-to-LDS when available), Kc (row-major + Kt
    // transposed from the in-flight registers), Vt (transposed slice).
#pragma unroll
    for (int j = 0; j < 8; j++) {
      int u = tid + j * 256, row = u >> 5, sg = (u & 31) * 8;
#if HAVE_ASYNC_LDS
      async_copy_b128(&qb[(size_t)(c0 + row) * DF + sg], &Qs[row * 264 + sg]);
#else
      int4 rq = *(const int4*)&qb[(size_t)(c0 + row) * DF + sg];
      *(int4*)&Qs[row * 264 + sg] = rq;
#endif
      int4 rk = *(const int4*)&kb[(size_t)(c0 + row) * DF + sg];
      *(int4*)&Ks[row * 264 + sg] = rk;
      const unsigned short* ek = (const unsigned short*)&rk;
#pragma unroll
      for (int jj = 0; jj < 8; jj++) Kt[(sg + jj) * 72 + row] = ek[jj];
    }
    {
      int c = tid >> 2, vseg = (tid & 3) * 16;
      int4 p0 = *(const int4*)&vb[(size_t)(c0 + c) * DV + vseg];
      int4 p1 = *(const int4*)&vb[(size_t)(c0 + c) * DV + vseg + 8];
      const unsigned short* e0 = (const unsigned short*)&p0;
      const unsigned short* e1 = (const unsigned short*)&p1;
#pragma unroll
      for (int j = 0; j < 8; j++) {
        Vt[(vseg + j) * 72 + c]     = e0[j];
        Vt[(vseg + 8 + j) * 72 + c] = e1[j];
      }
    }
    ASYNC_WAIT();
    __syncthreads();

    // Phase A: attn^T tiles (rt, ct0..ct0+1), K=256; shared A per k-step
    {
      F32x8 acc[2];
#pragma unroll
      for (int q = 0; q < 2; q++)
#pragma unroll
        for (int r = 0; r < 8; r++) acc[q].f[r] = 0.f;
#pragma unroll
      for (int ks = 0; ks < 8; ks++) {
        v16bf a = load_A(Ks, 264, rt * 16, ks * 32, lane);       // K rows
#pragma unroll
        for (int q = 0; q < 2; q++) {
          v16bf b = load_B_trans(Qs, 264, (ct0 + q) * 16, ks * 32, lane);
          acc[q].v = wmma_bf16(a, b, acc[q].v);
        }
      }
      int ebase = rt * 16 + 8 * half;       // score row base (contiguous)
#pragma unroll
      for (int q = 0; q < 2; q++) {
        int c = (ct0 + q) * 16 + nn;        // token column (fixed per lane)
        unsigned int pk[4];
#pragma unroll
        for (int rr = 0; rr < 4; rr++) {
          float lo = (c >= ebase + 2 * rr)     ? acc[q].f[2 * rr]     : 0.f;
          float hi = (c >= ebase + 2 * rr + 1) ? acc[q].f[2 * rr + 1] : 0.f;
          pk[rr] = pack_bf16(lo, hi);
        }
        *(int4*)&Asc[c * 72 + ebase] = make_int4(pk[0], pk[1], pk[2], pk[3]);
      }
    }
    __syncthreads();

    // Phase B: O^T tiles (rt, ct0..ct0+1) = V^T attn^T + S^T Q^T
    {
      F32x8 acc[2];
#pragma unroll
      for (int q = 0; q < 2; q++)
#pragma unroll
        for (int r = 0; r < 8; r++) acc[q].f[r] = 0.f;
#pragma unroll
      for (int ks = 0; ks < 2; ks++) {            // intra, K=64 (e)
        v16bf a = load_A(Vt, 72, rt * 16, ks * 32, lane);
#pragma unroll
        for (int q = 0; q < 2; q++) {
          v16bf b = load_B_trans(Asc, 72, (ct0 + q) * 16, ks * 32, lane);
          acc[q].v = wmma_bf16(a, b, acc[q].v);
        }
      }
#pragma unroll
      for (int ks = 0; ks < 8; ks++) {            // inter, K=256 (d)
        v16bf a = load_A(Sbt, 264, rt * 16, ks * 32, lane);
#pragma unroll
        for (int q = 0; q < 2; q++) {
          v16bf b = load_B_trans(Qs, 264, (ct0 + q) * 16, ks * 32, lane);
          acc[q].v = wmma_bf16(a, b, acc[q].v);
        }
      }
      int vbase = rt * 16 + 8 * half;
#pragma unroll
      for (int q = 0; q < 2; q++) {
        int c = (ct0 + q) * 16 + nn;
        float* dst = ob + (size_t)(c0 + c) * DV + vbase;
        *(float4*)dst       = make_float4(acc[q].f[0], acc[q].f[1], acc[q].f[2], acc[q].f[3]);
        *(float4*)(dst + 4) = make_float4(acc[q].f[4], acc[q].f[5], acc[q].f[6], acc[q].f[7]);
      }
    }
    __syncthreads();

    // Phase C: S += Kc^T @ Vc in registers; packed bf16 mirror Sbt[v][d].
    // Share B across both d-tiles and A across all four v-tiles.
#pragma unroll
    for (int ks = 0; ks < 2; ks++) {
      v16bf bv[4];
#pragma unroll
      for (int nt = 0; nt < 4; nt++)
        bv[nt] = load_B_trans(Vt, 72, nt * 16, ks * 32, lane);
#pragma unroll
      for (int d2 = 0; d2 < 2; d2++) {
        v16bf a = load_A(Kt, 72, (2 * wid + d2) * 16, ks * 32, lane);
#pragma unroll
        for (int nt = 0; nt < 4; nt++)
          Sreg[d2 * 4 + nt].v = wmma_bf16(a, bv[nt], Sreg[d2 * 4 + nt].v);
      }
    }
#pragma unroll
    for (int i = 0; i < 8; i++) {
      int dt = 2 * wid + (i >> 2), nt = i & 3;
      int col = nt * 16 + nn;               // v index (fixed per lane)
      int dbase = dt * 16 + 8 * half;       // d rows (contiguous)
      unsigned int pk[4];
#pragma unroll
      for (int rr = 0; rr < 4; rr++)
        pk[rr] = pack_bf16(Sreg[i].f[2 * rr], Sreg[i].f[2 * rr + 1]);
      *(int4*)&Sbt[col * 264 + dbase] = make_int4(pk[0], pk[1], pk[2], pk[3]);
    }
    __syncthreads();
  }
}

// ---------------------------------------------------------------------------
// RMSNorm over DV per (b,h,n) row; writes bf16 in [B, N, H*DV] for Wo GEMM.
// One wave per row; 8 rows per block.
// ---------------------------------------------------------------------------
__global__ __launch_bounds__(256) void rms_kernel(
    const float* __restrict__ Oacc, const float* __restrict__ rms_w,
    unsigned short* __restrict__ On) {
  const int lane = threadIdx.x & 31, wid = threadIdx.x >> 5;
  const size_t row = (size_t)blockIdx.x * 8 + wid;    // over B*H*N
  const int bh = (int)(row >> 11), ntok = (int)(row & (NN - 1));
  const int b = bh >> 3, h = bh & 7;
  const float* src = Oacc + row * DV;
  float vals[4], s = 0.f;
#pragma unroll
  for (int j = 0; j < 4; j++) { vals[j] = src[lane * 4 + j]; s += vals[j] * vals[j]; }
  s += __shfl_xor(s, 1, 32);
  s += __shfl_xor(s, 2, 32);
  s += __shfl_xor(s, 4, 32);
  s += __shfl_xor(s, 8, 32);
  s += __shfl_xor(s, 16, 32);
  float inv = rsqrtf(s * (1.0f / DV) + 1e-5f);
  unsigned short* dst = On + ((size_t)b * NN + ntok) * DD + h * DV + lane * 4;
  unsigned int w0 = pack_bf16(vals[0] * inv * rms_w[lane * 4 + 0],
                              vals[1] * inv * rms_w[lane * 4 + 1]);
  unsigned int w1 = pack_bf16(vals[2] * inv * rms_w[lane * 4 + 2],
                              vals[3] * inv * rms_w[lane * 4 + 3]);
  *(uint2*)dst = make_uint2(w0, w1);
}

// ---------------------------------------------------------------------------
// Launcher
// ---------------------------------------------------------------------------
extern "C" void kernel_launch(void* const* d_in, const int* in_sizes, int n_in,
                              void* d_out, int out_size, void* d_ws, size_t ws_size,
                              hipStream_t stream) {
  const float* x     = (const float*)d_in[0];
  const float* Wq    = (const float*)d_in[1];
  const float* Wk    = (const float*)d_in[2];
  const float* Wv    = (const float*)d_in[3];
  const float* Wo    = (const float*)d_in[4];
  const float* fmq_w = (const float*)d_in[5];
  const float* fmq_b = (const float*)d_in[6];
  const float* fmk_w = (const float*)d_in[7];
  const float* fmk_b = (const float*)d_in[8];
  const float* rms_w = (const float*)d_in[9];
  float* out = (float*)d_out;

  char* ws = (char*)d_ws;
  size_t off = 0;
  auto alloc = [&](size_t bytes) { char* p = ws + off; off += (bytes + 255) & ~(size_t)255; return p; };
  unsigned short* Xbf  = (unsigned short*)alloc((size_t)BN * DD * 2);
  unsigned short* Wqt  = (unsigned short*)alloc((size_t)DD * DD * 2);
  unsigned short* Wkt  = (unsigned short*)alloc((size_t)DD * DD * 2);
  unsigned short* Wvb  = (unsigned short*)alloc((size_t)DD * DD * 2);
  unsigned short* Wob  = (unsigned short*)alloc((size_t)DD * DD * 2);
  unsigned short* PhiQ = (unsigned short*)alloc((size_t)BB * HH * NN * DF * 2);
  unsigned short* PhiK = (unsigned short*)alloc((size_t)BB * HH * NN * DF * 2);
  unsigned short* Vh   = (unsigned short*)alloc((size_t)BB * HH * NN * DV * 2);
  float*          Oacc = (float*)alloc((size_t)BB * HH * NN * DV * 4);
  unsigned short* On   = (unsigned short*)alloc((size_t)BN * DD * 2);

  // 1) precision prep
  cvt_bf16_kernel<<<4096, 256, 0, stream>>>(x, (unsigned int*)Xbf, BN * DD / 2);
  make_wtilde_kernel<<<2048, 256, 0, stream>>>(fmq_w, Wq, Wqt);
  make_wtilde_kernel<<<2048, 256, 0, stream>>>(fmk_w, Wk, Wkt);
  cvt_bf16_kernel<<<1024, 256, 0, stream>>>(Wv, (unsigned int*)Wvb, DD * DD / 2);
  cvt_bf16_kernel<<<1024, 256, 0, stream>>>(Wo, (unsigned int*)Wob, DD * DD / 2);

  // 2) projections + hedgehog feature maps (scale = DF^-0.5 folded into phiQ)
  dim3 gg(BN / 128, DD / 128);
  const float scale = 0.0625f;  // 256^-0.5
  gemm_bf16_kernel<0><<<gg, 256, 0, stream>>>(Xbf, Wqt, fmq_b, PhiQ, nullptr, scale);
  gemm_bf16_kernel<1><<<gg, 256, 0, stream>>>(Xbf, Wkt, fmk_b, PhiK, nullptr, 1.0f);
  gemm_bf16_kernel<2><<<gg, 256, 0, stream>>>(Xbf, Wvb, nullptr, Vh, nullptr, 1.0f);

  // 3) chunked causal linear attention (state in VGPRs)
  attn_kernel<<<dim3(BB * HH, DV / DVS), 256, 0, stream>>>(PhiQ, PhiK, Vh, Oacc);

  // 4) RMSNorm -> bf16 [B,N,D]
  rms_kernel<<<(BB * HH * NN) / 8, 256, 0, stream>>>(Oacc, rms_w, On);

  // 5) output projection -> fp32 d_out
  gemm_bf16_kernel<3><<<gg, 256, 0, stream>>>(On, Wob, nullptr, nullptr, out, 1.0f);
}
